// SimpleGCN_65481071401041
// MI455X (gfx1250) — matrix operand order repaired and checked by reading, not compiled, not run on hardware
//
#include <hip/hip_runtime.h>
#include <hip/hip_bf16.h>

// ---------------------------------------------------------------------------
// SimpleGCN on MI455X (gfx1250, wave32, WMMA).
//   d   = rsqrt(sum(adj,1)+1e-8)
//   y1  = d ⊙ (adj @ (d ⊙ x))            [8192,128]   bf16 WMMA, fp32 acc
//   h   = relu(y1 @ W1^T + b1)           [8192,256]
//   y2  = d ⊙ (adj @ (d ⊙ h))            [8192,256]   bf16 WMMA, fp32 acc
//   out = y2 @ W2^T + b2                 [8192,64]
//
// adj (256 MB fp32) is streamed exactly 3x (deg + two GEMM passes) ~= 33 us
// at 23.3 TB/s. GEMM B-operands are staged in LDS (double-buffered) and the
// whole inner loop is software-pipelined: A global loads one k-step ahead,
// B ds_loads one N-tile ahead of each WMMA.
// ---------------------------------------------------------------------------

#define NROW 8192
#define NIN  128
#define NHID 256
#define NOUT 64

typedef __attribute__((ext_vector_type(16))) __bf16 v16bf;
typedef __attribute__((ext_vector_type(8)))  __bf16 v8bf;
typedef __attribute__((ext_vector_type(8)))  float  v8f;

// 32-byte (16 bf16) chunk helpers using only 16B-aligned ops.
__device__ inline v16bf load_chunk(const void* p) {
  const v8bf* q = (const v8bf*)p;
  const v8bf lo = q[0], hi = q[1];
  v16bf r;
#pragma unroll
  for (int i = 0; i < 8; ++i) { r[i] = lo[i]; r[i + 8] = hi[i]; }
  return r;
}
__device__ inline void store_chunk(void* p, v16bf v) {
  v8bf lo, hi;
#pragma unroll
  for (int i = 0; i < 8; ++i) { lo[i] = v[i]; hi[i] = v[i + 8]; }
  ((v8bf*)p)[0] = lo; ((v8bf*)p)[1] = hi;
}

// --------------------------- 1) degree / d ---------------------------------
__global__ __launch_bounds__(256) void deg_kernel(const float* __restrict__ adj,
                                                  float* __restrict__ dvec) {
  __shared__ float red[256];
  const int r = blockIdx.x;
  const float* row = adj + (size_t)r * NROW;
  float s = 0.f;
  for (int j = threadIdx.x * 4; j < NROW; j += 256 * 4) {
    const float4 v = *(const float4*)(row + j);
    s += v.x + v.y + v.z + v.w;
  }
  red[threadIdx.x] = s;
  __syncthreads();
  for (int off = 128; off > 0; off >>= 1) {
    if (threadIdx.x < off) red[threadIdx.x] += red[threadIdx.x + off];
    __syncthreads();
  }
  if (threadIdx.x == 0) dvec[r] = rsqrtf(red[0] + 1e-8f);
}

// ------------------ 2) xsT[c][r] = bf16(d[r]*x[r][c]) ----------------------
__global__ __launch_bounds__(256) void prep_xT(const float* __restrict__ x,
                                               const float* __restrict__ dvec,
                                               __bf16* __restrict__ xsT) {
  __shared__ float tile[32][33];
  const int rb = blockIdx.x * 32;
  const int cb = blockIdx.y * 32;
  const int tx = threadIdx.x & 31;
  const int ty = threadIdx.x >> 5;
#pragma unroll
  for (int i = 0; i < 32; i += 8) {
    const int r = rb + ty + i;
    tile[ty + i][tx] = dvec[r] * x[(size_t)r * NIN + cb + tx];
  }
  __syncthreads();
#pragma unroll
  for (int i = 0; i < 32; i += 8) {
    const int c = cb + ty + i;
    xsT[(size_t)c * NROW + rb + tx] = (__bf16)tile[tx][ty + i];
  }
}

// --------- 3)/5) C[r][n] = d[r] * sum_k adj[r][k] * BT[n][k]  (WMMA) -------
// A:  adj fp32 [8192,8192], converted fp32->bf16 in-register (v_cvt_pk_bf16).
// BT: bf16 [NCOLS,8192] row-major (pre-scaled by d along k).
// Block = 4 waves, M-tile = 64 rows, full NCOLS per block (adj streamed 1x).
// Pipelining: A global loads one k-step ahead (ping-pong via unroll 2);
// B k-slab staged in LDS double-buffered; B ds_load one tile ahead of WMMA.
template <int NCOLS>
__global__ __launch_bounds__(128) void gcn_adj_gemm(const float* __restrict__ A,
                                                    const __bf16* __restrict__ BT,
                                                    const float* __restrict__ dvec,
                                                    float* __restrict__ C) {
  constexpr int NT     = NCOLS / 16;      // N tiles per wave (8 or 16)
  constexpr int CHUNKS = 2 * NCOLS;       // 32B chunks per k-slab
  constexpr int STRIDE = 48;              // 32B data + 16B pad (16B aligned)
  constexpr int CPT    = CHUNKS / 128;    // chunks staged per thread (2 or 4)
  __shared__ alignas(16) char smem[2 * CHUNKS * STRIDE];

  const int tid   = threadIdx.x;
  const int lane  = tid & 31;
  const int wave  = tid >> 5;                       // 0..3
  const int row0  = blockIdx.x * 64 + wave * 16;    // wave's 16 output rows
  const int lrow  = lane & 15;                      // fragment row / column
  const int khalf = lane >> 4;  // 0: K{0..15}; 1: K{16..31}

  const float* arow = A + (size_t)(row0 + lrow) * NROW + khalf * 16;

  // ---- stage B k-slab 0 into LDS buffer 0 ----
#pragma unroll
  for (int i = 0; i < CPT; ++i) {
    const int ch = tid + i * 128;        // ch = h*NCOLS + c
    const int c  = ch % NCOLS;
    const int h  = ch / NCOLS;
    const v16bf v = load_chunk(BT + (size_t)c * NROW + h * 16);
    store_chunk(smem + (size_t)ch * STRIDE, v);
  }

  // ---- preload A fragment raws for k0 = 0 ----
  float4 an0 = *(const float4*)(arow);
  float4 an1 = *(const float4*)(arow + 4);
  float4 an2 = *(const float4*)(arow + 8);
  float4 an3 = *(const float4*)(arow + 12);

  __syncthreads();

  v8f acc[NT] = {};

#pragma unroll 2
  for (int k0 = 0; k0 < NROW; k0 += 32) {
    const int  cur  = (k0 >> 5) & 1;
    const bool more = (k0 + 32) < NROW;

    // consume last iteration's A registers; issue next iteration's A loads
    const float4 a0 = an0, a1 = an1, a2 = an2, a3 = an3;

    // ---- issue NEXT B slab's global loads (consumed by ds_store below) ----
    v16bf st[CPT];
    if (more) {
#pragma unroll
      for (int i = 0; i < CPT; ++i) {
        const int ch = tid + i * 128;
        const int c  = ch % NCOLS;
        const int h  = ch / NCOLS;
        st[i] = load_chunk(BT + (size_t)c * NROW + (k0 + 32) + h * 16);
      }
      // ---- issue NEXT A fragment loads (consumed next iteration) ----
      const float* ap = arow + k0 + 32;
      __builtin_prefetch(ap + 32, 0, 1);  // global_prefetch_b8: 2 steps ahead
      an0 = *(const float4*)(ap);
      an1 = *(const float4*)(ap + 4);
      an2 = *(const float4*)(ap + 8);
      an3 = *(const float4*)(ap + 12);
    }

    // ---- A fragment: 16x32 fp32 -> bf16 (contiguous k = 16*khalf + s, same order as B) ----
    v16bf afrag;
    afrag[0]  = (__bf16)a0.x; afrag[1]  = (__bf16)a0.y;
    afrag[2]  = (__bf16)a0.z; afrag[3]  = (__bf16)a0.w;
    afrag[4]  = (__bf16)a1.x; afrag[5]  = (__bf16)a1.y;
    afrag[6]  = (__bf16)a1.z; afrag[7]  = (__bf16)a1.w;
    afrag[8]  = (__bf16)a2.x; afrag[9]  = (__bf16)a2.y;
    afrag[10] = (__bf16)a2.z; afrag[11] = (__bf16)a2.w;
    afrag[12] = (__bf16)a3.x; afrag[13] = (__bf16)a3.y;
    afrag[14] = (__bf16)a3.z; afrag[15] = (__bf16)a3.w;

    // ---- WMMA chain; B ds_load pipelined one tile ahead ----
    const char* bufc = smem + (size_t)cur * CHUNKS * STRIDE;
    const char* bbase = bufc + (size_t)(khalf * NCOLS + lrow) * STRIDE;
    v16bf bcur = load_chunk(bbase);
#pragma unroll
    for (int nt = 0; nt < NT; ++nt) {
      const v16bf bthis = bcur;
      if (nt + 1 < NT) bcur = load_chunk(bbase + (size_t)(nt + 1) * 16 * STRIDE);
      acc[nt] = __builtin_amdgcn_wmma_f32_16x16x32_bf16(
          /*neg_a=*/false, afrag, /*neg_b=*/false, bthis,
          /*c_mod=*/(short)0, acc[nt], /*reuse_a=*/false, /*reuse_b=*/false);
    }

    // ---- commit next slab into the other buffer ----
    if (more) {
      char* bufn = smem + (size_t)(cur ^ 1) * CHUNKS * STRIDE;
#pragma unroll
      for (int i = 0; i < CPT; ++i)
        store_chunk(bufn + (size_t)(tid + i * 128) * STRIDE, st[i]);
    }
    __syncthreads();
  }

  // ---- epilogue: fold d[r]; C/D layout: VGPR i -> M=i (lo) / M=i+8 (hi) ----
  const int rbase = row0 + (khalf ? 8 : 0);
  const int cbase = lrow;
#pragma unroll
  for (int nt = 0; nt < NT; ++nt) {
#pragma unroll
    for (int i = 0; i < 8; ++i) {
      const int r = rbase + i;
      C[(size_t)r * NCOLS + cbase + nt * 16] = acc[nt][i] * dvec[r];
    }
  }
}

// -------- 4) hsT[o][r] = bf16(d[r]*relu(y1[r]·W1[o] + b1[o])) --------------
__global__ __launch_bounds__(256) void mlp1_kernel(const float* __restrict__ y1,
                                                   const float* __restrict__ W1,
                                                   const float* __restrict__ b1,
                                                   const float* __restrict__ dvec,
                                                   __bf16* __restrict__ hsT) {
  __shared__ float xrow[NIN];
  const int r = blockIdx.x;
  if (threadIdx.x < NIN) xrow[threadIdx.x] = y1[(size_t)r * NIN + threadIdx.x];
  __syncthreads();
  const int o = threadIdx.x;  // 0..255
  const float* w = W1 + (size_t)o * NIN;
  float acc = b1[o];
#pragma unroll 8
  for (int c = 0; c < NIN; ++c) acc = fmaf(xrow[c], w[c], acc);
  acc = fmaxf(acc, 0.f);
  hsT[(size_t)o * NROW + r] = (__bf16)(acc * dvec[r]);
}

// ----------------- 6) out[r][o] = y2[r]·W2[o] + b2[o] ----------------------
__global__ __launch_bounds__(256) void mlp2_kernel(const float* __restrict__ y2,
                                                   const float* __restrict__ W2,
                                                   const float* __restrict__ b2,
                                                   float* __restrict__ out) {
  __shared__ float yrow[4][NHID];
  const int r0 = blockIdx.x * 4;
  for (int i = threadIdx.x; i < 4 * NHID; i += 256)
    yrow[i >> 8][i & 255] = y2[(size_t)r0 * NHID + i];
  __syncthreads();
  const int rr = threadIdx.x >> 6;  // 0..3
  const int o  = threadIdx.x & 63;  // 0..63
  const float* w = W2 + (size_t)o * NHID;
  float acc = b2[o];
#pragma unroll 8
  for (int c = 0; c < NHID; ++c) acc = fmaf(yrow[rr][c], w[c], acc);
  out[(size_t)(r0 + rr) * NOUT + o] = acc;
}

// ---------------------------------------------------------------------------
extern "C" void kernel_launch(void* const* d_in, const int* in_sizes, int n_in,
                              void* d_out, int out_size, void* d_ws, size_t ws_size,
                              hipStream_t stream) {
  const float* x   = (const float*)d_in[0];  // [8192,128]
  const float* adj = (const float*)d_in[1];  // [8192,8192]
  const float* W1  = (const float*)d_in[2];  // [256,128]
  const float* b1  = (const float*)d_in[3];  // [256]
  const float* W2  = (const float*)d_in[4];  // [64,256]
  const float* b2  = (const float*)d_in[5];  // [64]
  float* out = (float*)d_out;                // [8192,64]

  // Workspace carve (all offsets 4 KB aligned; total ~19 MB)
  char* ws = (char*)d_ws;
  float*  dvec = (float*)(ws);                                  // 32 KB
  __bf16* xsT  = (__bf16*)(ws + 32768);                         // 2 MB  [128,8192]
  float*  y1   = (float*)(ws + 32768 + 2097152);                // 4 MB  [8192,128]
  __bf16* hsT  = (__bf16*)(ws + 32768 + 2097152 + 4194304);     // 4 MB  [256,8192]
  float*  y2   = (float*)(ws + 32768 + 2097152 + 8388608);      // 8 MB  [8192,256]

  deg_kernel<<<NROW, 256, 0, stream>>>(adj, dvec);
  prep_xT<<<dim3(NROW / 32, NIN / 32), 256, 0, stream>>>(x, dvec, xsT);
  gcn_adj_gemm<NIN><<<NROW / 64, 128, 0, stream>>>(adj, xsT, dvec, y1);
  mlp1_kernel<<<NROW, 256, 0, stream>>>(y1, W1, b1, dvec, hsT);
  gcn_adj_gemm<NHID><<<NROW / 64, 128, 0, stream>>>(adj, hsT, dvec, y2);
  mlp2_kernel<<<NROW / 4, 256, 0, stream>>>(y2, W2, b2, out);
}
